// FeatureBasedEdgeBuilder_2946347565057
// MI455X (gfx1250) — matrix operand order repaired and checked by reading, not compiled, not run on hardware
//
#include <hip/hip_runtime.h>
#include <hip/hip_bf16.h>

typedef __attribute__((ext_vector_type(16))) __bf16 v16bf;
typedef __attribute__((ext_vector_type(8)))  __bf16 v8bf;
typedef __attribute__((ext_vector_type(8)))  float  v8f;

#define BB 4
#define SS 2048
#define DD 1024
#define MM (BB * SS)     // 8192 total rows
#define TWAVE (SS / 4)   // 512 score columns per wave in scores_topk

union Frag16 { v16bf v; v8bf h[2]; };

__device__ __forceinline__ v8bf load8bf(const __bf16* p) {
  return *reinterpret_cast<const v8bf*>(p);
}

__device__ __forceinline__ v8f wmma_bf16(const Frag16& a, const Frag16& b, v8f c) {
  return __builtin_amdgcn_wmma_f32_16x16x32_bf16(false, a.v, false, b.v,
                                                 (short)0, c, false, false);
}

// ---------------------------------------------------------------------------
// Kernel 0: fp32 -> bf16 convert
// ---------------------------------------------------------------------------
__global__ void __launch_bounds__(256)
cvt_f32_bf16(const float* __restrict__ in, __bf16* __restrict__ out, int n) {
  int i = blockIdx.x * blockDim.x + threadIdx.x;
  if (i < n) out[i] = (__bf16)in[i];
}

// ---------------------------------------------------------------------------
// Kernel 1: dual projection GEMM  phi = H @ Wphi^T, psi = H @ Wpsi^T
// One wave: 32 rows x 16 cols of BOTH outputs. Per K-step: 4 b128-pair loads
// feed 4 WMMAs (A frags shared by phi/psi, B frags shared by both row tiles).
// ---------------------------------------------------------------------------
__global__ void __launch_bounds__(128)
proj_gemm(const __bf16* __restrict__ hB,
          const __bf16* __restrict__ wPhi,
          const __bf16* __restrict__ wPsi,
          __bf16* __restrict__ phi,
          __bf16* __restrict__ psi) {
  const int wave = threadIdx.x >> 5;
  const int lane = threadIdx.x & 31;
  const int tile = blockIdx.x * 4 + wave;
  const int nTilesN = DD / 16;                 // 64
  const int row0 = (tile / nTilesN) * 32;
  const int col0 = (tile % nTilesN) * 16;
  const int l15  = lane & 15;
  const int kh   = lane >> 4;

  const __bf16* aRow0 = hB   + (size_t)(row0 + l15) * DD;
  const __bf16* aRow1 = hB   + (size_t)(row0 + 16 + l15) * DD;
  const __bf16* bRowP = wPhi + (size_t)(col0 + l15) * DD;
  const __bf16* bRowQ = wPsi + (size_t)(col0 + l15) * DD;

  v8f accP0 = {}, accP1 = {}, accQ0 = {}, accQ1 = {};
  for (int k0 = 0; k0 < DD; k0 += 32) {
    Frag16 a0, a1, bp, bq;
    // A 16x32 bf16 layout (ISA 7.12.2): two contiguous 8-elem K runs per lane.
    a0.h[0] = load8bf(aRow0 + k0 + kh * 8);
    a0.h[1] = load8bf(aRow0 + k0 + 16 + kh * 8);
    a1.h[0] = load8bf(aRow1 + k0 + kh * 8);
    a1.h[1] = load8bf(aRow1 + k0 + 16 + kh * 8);
    // B 32x16 bf16 layout: same K order as A (two 8-elem K runs per lane).
    bp.h[0] = load8bf(bRowP + k0 + kh * 8);
    bp.h[1] = load8bf(bRowP + k0 + 16 + kh * 8);
    bq.h[0] = load8bf(bRowQ + k0 + kh * 8);
    bq.h[1] = load8bf(bRowQ + k0 + 16 + kh * 8);
    accP0 = wmma_bf16(a0, bp, accP0);
    accQ0 = wmma_bf16(a0, bq, accQ0);
    accP1 = wmma_bf16(a1, bp, accP1);
    accQ1 = wmma_bf16(a1, bq, accQ1);
  }
  // C/D layout: VGPR r -> M = r + 8*kh, N = lane&15
  #pragma unroll
  for (int r = 0; r < 8; ++r) {
    int m0 = row0 + r + 8 * kh;
    int m1 = m0 + 16;
    phi[(size_t)m0 * DD + col0 + l15] = (__bf16)accP0[r];
    psi[(size_t)m0 * DD + col0 + l15] = (__bf16)accQ0[r];
    phi[(size_t)m1 * DD + col0 + l15] = (__bf16)accP1[r];
    psi[(size_t)m1 * DD + col0 + l15] = (__bf16)accQ1[r];
  }
}

// ---------------------------------------------------------------------------
// Kernel 2: fused scores GEMM + exp + running top-8 per row.
// Block = 32-row strip. Each of the 4 waves covers 512 t-columns with 2x2
// register blocking (4 WMMAs per 4 fragment loads); per-wave top-8 kept in
// registers (one row per lane), merged across waves through LDS at the end.
// ---------------------------------------------------------------------------
__global__ void __launch_bounds__(128)
scores_topk(const __bf16* __restrict__ phi,
            const __bf16* __restrict__ psi,
            float* __restrict__ topv,
            int*   __restrict__ topi,
            int*   __restrict__ cnt) {
  __shared__ float tile[4][32][33];   // per-wave exp(score) tile (+pad)
  __shared__ float mv[4][32][8];      // per-wave top-8 values
  __shared__ int   mi[4][32][8];      // per-wave top-8 indices
  const int wave = threadIdx.x >> 5;
  const int lane = threadIdx.x & 31;
  const int row0 = blockIdx.x * 32;
  const int rb   = (row0 / SS) * SS;  // batch base (32 | SS)
  const int l15  = lane & 15;
  const int kh   = lane >> 4;
  const int tBase = wave * TWAVE;

  const __bf16* aRow0 = phi + (size_t)(row0 + l15) * DD;
  const __bf16* aRow1 = phi + (size_t)(row0 + 16 + l15) * DD;

  float tv[8];
  int   ti[8];
  #pragma unroll
  for (int j = 0; j < 8; ++j) { tv[j] = -1.0f; ti[j] = 0; }  // exp() > 0

  for (int tt = 0; tt < TWAVE; tt += 32) {
    const int t0 = tBase + tt;
    const __bf16* bRow0 = psi + (size_t)(rb + t0 + l15) * DD;
    const __bf16* bRow1 = psi + (size_t)(rb + t0 + 16 + l15) * DD;
    // prefetch next t-tile's psi rows (resident in L2; covers latency)
    __builtin_prefetch(bRow0 + (size_t)32 * DD, 0, 1);
    __builtin_prefetch(bRow1 + (size_t)32 * DD, 0, 1);

    v8f a00 = {}, a01 = {}, a10 = {}, a11 = {};
    for (int k0 = 0; k0 < DD; k0 += 32) {
      Frag16 fa0, fa1, fb0, fb1;
      fa0.h[0] = load8bf(aRow0 + k0 + kh * 8);
      fa0.h[1] = load8bf(aRow0 + k0 + 16 + kh * 8);
      fa1.h[0] = load8bf(aRow1 + k0 + kh * 8);
      fa1.h[1] = load8bf(aRow1 + k0 + 16 + kh * 8);
      fb0.h[0] = load8bf(bRow0 + k0 + kh * 8);
      fb0.h[1] = load8bf(bRow0 + k0 + 16 + kh * 8);
      fb1.h[0] = load8bf(bRow1 + k0 + kh * 8);
      fb1.h[1] = load8bf(bRow1 + k0 + 16 + kh * 8);
      a00 = wmma_bf16(fa0, fb0, a00);
      a01 = wmma_bf16(fa0, fb1, a01);
      a10 = wmma_bf16(fa1, fb0, a10);
      a11 = wmma_bf16(fa1, fb1, a11);
    }
    #pragma unroll
    for (int r = 0; r < 8; ++r) {
      int rr = r + 8 * kh;
      tile[wave][rr][l15]           = __expf(a00[r]);
      tile[wave][rr][16 + l15]      = __expf(a01[r]);
      tile[wave][16 + rr][l15]      = __expf(a10[r]);
      tile[wave][16 + rr][16 + l15] = __expf(a11[r]);
    }
    __syncthreads();                 // uniform: all waves run TWAVE/32 iters
    // one row per lane; scan 32 candidates, keep tv sorted ascending
    #pragma unroll
    for (int j = 0; j < 32; ++j) {
      float v = tile[wave][lane][j];
      if (v > tv[0]) {
        tv[0] = v; ti[0] = t0 + j;
        #pragma unroll
        for (int p = 0; p < 7; ++p) {
          if (tv[p] > tv[p + 1]) {
            float fv = tv[p]; tv[p] = tv[p + 1]; tv[p + 1] = fv;
            int   iv = ti[p]; ti[p] = ti[p + 1]; ti[p + 1] = iv;
          }
        }
      }
    }
    __syncthreads();
  }

  // publish per-wave top-8, then wave 0 merges the 4 partial lists per row
  #pragma unroll
  for (int j = 0; j < 8; ++j) { mv[wave][lane][j] = tv[j]; mi[wave][lane][j] = ti[j]; }
  __syncthreads();

  if (wave == 0) {
    float fv[8]; int fi[8];
    #pragma unroll
    for (int j = 0; j < 8; ++j) { fv[j] = -1.0f; fi[j] = 0; }
    for (int w = 0; w < 4; ++w) {
      #pragma unroll
      for (int j = 0; j < 8; ++j) {
        float v = mv[w][lane][j];
        if (v > fv[0]) {
          fv[0] = v; fi[0] = mi[w][lane][j];
          #pragma unroll
          for (int p = 0; p < 7; ++p) {
            if (fv[p] > fv[p + 1]) {
              float t = fv[p]; fv[p] = fv[p + 1]; fv[p + 1] = t;
              int  it = fi[p]; fi[p] = fi[p + 1]; fi[p + 1] = it;
            }
          }
        }
      }
    }
    int g = row0 + lane;
    int sLocal = g - rb;
    float sum = 1e-8f;
    #pragma unroll
    for (int j = 0; j < 8; ++j) sum += fv[j];
    float inv = 1.0f / sum;
    int c = 0;
    #pragma unroll
    for (int j = 0; j < 8; ++j) {    // descending order (matches top_k)
      float v = fv[7 - j] * inv;
      int   t = fi[7 - j];
      topv[g * 8 + j] = v;
      topi[g * 8 + j] = t;
      if (t != sLocal) ++c;
    }
    cnt[g] = c;
  }
}

// ---------------------------------------------------------------------------
// Kernel 3: single-block exclusive scan of per-row edge counts (M = 8192)
// ---------------------------------------------------------------------------
__global__ void __launch_bounds__(256)
scan_counts(const int* __restrict__ cnt, int* __restrict__ offs,
            int* __restrict__ total) {
  __shared__ int partial[256];
  const int t = threadIdx.x;
  const int base = t * (MM / 256);             // 32 rows per thread
  int local[32];
  int run = 0;
  #pragma unroll
  for (int i = 0; i < 32; ++i) { local[i] = run; run += cnt[base + i]; }
  partial[t] = run;
  __syncthreads();
  for (int d = 1; d < 256; d <<= 1) {          // Hillis-Steele inclusive scan
    int v = (t >= d) ? partial[t - d] : 0;
    __syncthreads();
    partial[t] += v;
    __syncthreads();
  }
  int excl = (t == 0) ? 0 : partial[t - 1];
  #pragma unroll
  for (int i = 0; i < 32; ++i) offs[base + i] = excl + local[i];
  if (t == 255) *total = partial[255];
}

// ---------------------------------------------------------------------------
// Kernel 4: deterministic compaction into [src(E) | dst(E) | weights(E)]
// ---------------------------------------------------------------------------
__global__ void __launch_bounds__(256)
emit_edges(const float* __restrict__ topv, const int* __restrict__ topi,
           const int* __restrict__ offs, const int* __restrict__ total,
           float* __restrict__ out) {
  int g = blockIdx.x * blockDim.x + threadIdx.x;
  if (g >= MM) return;
  int E = *total;
  int rb = (g / SS) * SS;
  int sLocal = g - rb;
  int o = offs[g];
  #pragma unroll
  for (int j = 0; j < 8; ++j) {
    int t = topi[g * 8 + j];
    if (t != sLocal) {                          // drop self-loops
      out[o]         = (float)g;                // src (global node id)
      out[E + o]     = (float)(rb + t);         // dst (global node id)
      out[2 * E + o] = topv[g * 8 + j];         // normalized weight
      ++o;
    }
  }
}

// ---------------------------------------------------------------------------
extern "C" void kernel_launch(void* const* d_in, const int* in_sizes, int n_in,
                              void* d_out, int out_size, void* d_ws, size_t ws_size,
                              hipStream_t stream) {
  (void)in_sizes; (void)n_in; (void)out_size; (void)ws_size;
  const float* H    = (const float*)d_in[0];
  const float* Wphi = (const float*)d_in[1];
  const float* Wpsi = (const float*)d_in[2];

  char* ws = (char*)d_ws;
  size_t off = 0;
  __bf16* hB    = (__bf16*)(ws + off); off += (size_t)MM * DD * 2;  // 16 MB
  __bf16* wPhiB = (__bf16*)(ws + off); off += (size_t)DD * DD * 2;  //  2 MB
  __bf16* wPsiB = (__bf16*)(ws + off); off += (size_t)DD * DD * 2;  //  2 MB
  __bf16* phiB  = (__bf16*)(ws + off); off += (size_t)MM * DD * 2;  // 16 MB
  __bf16* psiB  = (__bf16*)(ws + off); off += (size_t)MM * DD * 2;  // 16 MB
  float*  topv  = (float*)(ws + off);  off += (size_t)MM * 8 * 4;
  int*    topi  = (int*)(ws + off);    off += (size_t)MM * 8 * 4;
  int*    cnt   = (int*)(ws + off);    off += (size_t)MM * 4;
  int*    offs  = (int*)(ws + off);    off += (size_t)MM * 4;
  int*    total = (int*)(ws + off);    off += 4;

  cvt_f32_bf16<<<(MM * DD) / 256, 256, 0, stream>>>(H, hB, MM * DD);
  cvt_f32_bf16<<<(DD * DD) / 256, 256, 0, stream>>>(Wphi, wPhiB, DD * DD);
  cvt_f32_bf16<<<(DD * DD) / 256, 256, 0, stream>>>(Wpsi, wPsiB, DD * DD);

  // (M/32)*(D/16) = 16384 wave-tiles, 4 waves per 128-thread block
  proj_gemm<<<(MM / 32) * (DD / 16) / 4, 128, 0, stream>>>(hB, wPhiB, wPsiB,
                                                           phiB, psiB);
  // one block per 32-row strip; 4 waves split the 2048 t-columns
  scores_topk<<<MM / 32, 128, 0, stream>>>(phiB, psiB, topv, topi, cnt);

  scan_counts<<<1, 256, 0, stream>>>(cnt, offs, total);
  emit_edges<<<MM / 256, 256, 0, stream>>>(topv, topi, offs, total,
                                           (float*)d_out);
}
